// Attention_23579370455635
// MI455X (gfx1250) — matrix operand order repaired and checked by reading, not compiled, not run on hardware
//
#include <hip/hip_runtime.h>

typedef __attribute__((ext_vector_type(16))) _Float16 v16h;
typedef __attribute__((ext_vector_type(8)))  _Float16 v8h;
typedef __attribute__((ext_vector_type(8)))  float    v8f;
typedef __attribute__((ext_vector_type(4)))  unsigned int v4u;
typedef __attribute__((ext_vector_type(8)))  int      v8i;
typedef __attribute__((ext_vector_type(4)))  int      v4i;

#define BATCH        2
#define NUM_HEADS    32
#define HEAD_DIM     128
#define NUM_KV_HEADS 8
#define SEQ          2048
#define QTILE        128
#define KVTILE       32
#define GQA          (NUM_HEADS / NUM_KV_HEADS)
#define VSTRIDE      (KVTILE + 8)   /* 40 halves = 80B row stride, keeps 16B align */
#define SCALE        0.08838834764831845f
#define NEG_INF      (-1e30f)

__device__ __forceinline__ v16h cat8h(v8h lo, v8h hi) {
  return __builtin_shufflevector(lo, hi, 0,1,2,3,4,5,6,7,8,9,10,11,12,13,14,15);
}

// Issue one TDM DMA: 32x128 fp32 tile, row stride = NUM_KV_HEADS*HEAD_DIM floats,
// global -> LDS at byte offset lds_off.  D# layout per CDNA5 ISA ch.8.
// This toolchain's builtin takes 6 args (g0, g1, g2, g3, g4, cpol).
__device__ __forceinline__ void tdm_load_tile_32x128_f32(const float* gsrc,
                                                         unsigned lds_off) {
  unsigned long long ga = (unsigned long long)(uintptr_t)gsrc;
  v4u g0 = { 1u,                                    // count=1 (valid user D#)
             lds_off,                               // lds_addr (bytes)
             (unsigned)ga,                          // global_addr[31:0]
             (unsigned)(ga >> 32) | (2u << 30) };   // global_addr[56:32] | type=2
  v8i g1 = { (int)0x00020000,                       // data_size = 4 bytes
             (int)(HEAD_DIM << 16),                 // tensor_dim0 = 128  (bits 79:48)
             (int)(SEQ << 16),                      // tensor_dim1 = 2048 (bits 111:80)
             (int)(HEAD_DIM << 16),                 // tile_dim0 = 128    (bits 127:112)
             (int)KVTILE,                           // tile_dim1 = 32
             (int)(NUM_KV_HEADS * HEAD_DIM),        // tensor_dim0_stride = 1024
             0, 0 };
  v4i gz4 = { 0, 0, 0, 0 };                         // 2D tile: groups 2/3 unused
  v8i gz8 = { 0, 0, 0, 0, 0, 0, 0, 0 };             // extra group (zero-filled)
  __builtin_amdgcn_tensor_load_to_lds(g0, g1, gz4, gz4, gz8, 0);
}

__global__ __launch_bounds__(256, 1) void fa2_gqa_causal(
    const float* __restrict__ Q, const float* __restrict__ K,
    const float* __restrict__ V, float* __restrict__ O) {
  __shared__ float    Kf[2][KVTILE][HEAD_DIM];   // TDM fp32 double buffer (32 KB)
  __shared__ float    Vf[2][KVTILE][HEAD_DIM];   // TDM fp32 double buffer (32 KB)
  __shared__ _Float16 Ks[KVTILE][HEAD_DIM];      // K tile f16, row-major  (8 KB)
  __shared__ _Float16 Vt[HEAD_DIM][VSTRIDE];     // V tile f16, transposed (10 KB)
  __shared__ _Float16 Pb[8][16][KVTILE];         // per-wave P scratch     (8 KB)

  const int tid  = threadIdx.x;
  const int wave = tid >> 5;
  const int lane = tid & 31;
  const int ln16 = lane & 15;
  const int sel  = lane >> 4;

  const int qtiles = SEQ / QTILE;                // 16
  const int blk = blockIdx.x;
  const int b   = blk / (NUM_HEADS * qtiles);
  const int rem = blk % (NUM_HEADS * qtiles);
  const int h   = rem / qtiles;
  const int qt  = rem % qtiles;
  const int hkv = h / GQA;

  const int q_base = qt * QTILE;
  const int q_row0 = q_base + wave * 16;

  const float* kbase = K + ((size_t)b * SEQ * NUM_KV_HEADS + hkv) * HEAD_DIM;
  const float* vbase = V + ((size_t)b * SEQ * NUM_KV_HEADS + hkv) * HEAD_DIM;
  const size_t rowstride = (size_t)NUM_KV_HEADS * HEAD_DIM;   // 1024 floats

  // ---- Q fragment in WMMA A-layout (f16), rows q_row0..+15, D=128 ----
  v16h qfrag[4];
  {
    const int qrow = q_row0 + ln16;
    const float* qp = Q + ((size_t)(b * SEQ + qrow) * NUM_HEADS + h) * HEAD_DIM;
    #pragma unroll
    for (int c = 0; c < 4; ++c) {
      union { v16h v; _Float16 e[16]; } u;
      const int d0 = c * 32 + sel * 8;
      #pragma unroll
      for (int i = 0; i < 8; ++i) {
        u.e[i]     = (_Float16)qp[d0 + i];        // K = d0 .. d0+7
        u.e[8 + i] = (_Float16)qp[d0 + 16 + i];   // K = d0+16 .. d0+23
      }
      qfrag[c] = u.v;
    }
  }

  // ---- kick off TDM for tile 0 (wave 0 only; TDM ignores EXEC, 1 issue/wave)
  if (wave == 0) {
    tdm_load_tile_32x128_f32(kbase, (unsigned)(uintptr_t)&Kf[0][0][0]);
    tdm_load_tile_32x128_f32(vbase, (unsigned)(uintptr_t)&Vf[0][0][0]);
  }

  // online-softmax state: per lane, 8 rows (M = sel*8 + r), columns = ln16
  float m_i[8], l_i[8];
  v8f   o_acc[8];
  #pragma unroll
  for (int r = 0; r < 8; ++r) { m_i[r] = NEG_INF; l_i[r] = 0.0f; }
  #pragma unroll
  for (int n = 0; n < 8; ++n) { v8f z = {}; o_acc[n] = z; }

  const int kv_end = q_base + QTILE;             // causal bound for this tile
  int nb = 0;
  for (int kv0 = 0; kv0 < kv_end; kv0 += KVTILE) {
    if (wave == 0) __builtin_amdgcn_s_wait_tensorcnt((short)0);
    __syncthreads();   // TDM data visible; prior iter's f16 readers done

    // ---- cooperative convert fp32 tile -> f16 tiles (once per block) ----
    {
      const int row  = tid >> 3;                 // 0..31
      const int colb = (tid & 7) * 16;           // 0..112
      const float4* kp4 = (const float4*)&Kf[nb][row][colb];
      const float4* vp4 = (const float4*)&Vf[nb][row][colb];
      #pragma unroll
      for (int i = 0; i < 4; ++i) {
        float4 kf = kp4[i];
        float4 vf = vp4[i];
        const int c = colb + i * 4;
        Ks[row][c + 0] = (_Float16)kf.x;
        Ks[row][c + 1] = (_Float16)kf.y;
        Ks[row][c + 2] = (_Float16)kf.z;
        Ks[row][c + 3] = (_Float16)kf.w;
        Vt[c + 0][row] = (_Float16)vf.x;
        Vt[c + 1][row] = (_Float16)vf.y;
        Vt[c + 2][row] = (_Float16)vf.z;
        Vt[c + 3][row] = (_Float16)vf.w;
      }
    }

    // ---- wave 0: launch DMA for next tile into the other buffer ----
    if (wave == 0 && kv0 + KVTILE < kv_end) {
      const size_t off = (size_t)(kv0 + KVTILE) * rowstride;
      tdm_load_tile_32x128_f32(kbase + off, (unsigned)(uintptr_t)&Kf[nb ^ 1][0][0]);
      tdm_load_tile_32x128_f32(vbase + off, (unsigned)(uintptr_t)&Vf[nb ^ 1][0][0]);
    }
    __syncthreads();   // f16 tiles ready for all waves

    // ---- preload all K fragments (distinct regs -> overlap DS with WMMA) ----
    v16h kfr[2][4];
    #pragma unroll
    for (int j = 0; j < 2; ++j) {
      #pragma unroll
      for (int c = 0; c < 4; ++c) {
        // B[k][n] = Ks[j*16+n][c*32+k]; per lane n=ln16, k=sel*8+s / 16+sel*8+(s-8) (table order)
        v8h lo = *(const v8h*)&Ks[j * 16 + ln16][c * 32 + sel * 8];
        v8h hi = *(const v8h*)&Ks[j * 16 + ln16][c * 32 + 16 + sel * 8];
        kfr[j][c] = cat8h(lo, hi);
      }
    }

    // ---- scores: S(16x32) = Q(16x128) . K^T ----
    v8f sacc[2];
    #pragma unroll
    for (int j = 0; j < 2; ++j) {
      v8f acc = {};
      #pragma unroll
      for (int c = 0; c < 4; ++c)
        acc = __builtin_amdgcn_wmma_f32_16x16x32_f16(
            false, qfrag[c], false, kfr[j][c], (short)0, acc, false, false);
      sacc[j] = acc;
    }

    // ---- causal mask + online softmax (8 rows per lane) ----
    #pragma unroll
    for (int r = 0; r < 8; ++r) {
      const int qi   = q_row0 + sel * 8 + r;
      const int kvi0 = kv0 + ln16;
      const int kvi1 = kv0 + 16 + ln16;
      float s0 = (kvi0 <= qi) ? sacc[0][r] * SCALE : NEG_INF;
      float s1 = (kvi1 <= qi) ? sacc[1][r] * SCALE : NEG_INF;
      float mx = fmaxf(s0, s1);
      #pragma unroll
      for (int off = 1; off < 16; off <<= 1)     // stays inside 16-lane half
        mx = fmaxf(mx, __shfl_xor(mx, off, 32));
      const float m_new = fmaxf(m_i[r], mx);
      const float alpha = __expf(m_i[r] - m_new);
      const float p0 = __expf(s0 - m_new);
      const float p1 = __expf(s1 - m_new);
      float ls = p0 + p1;
      #pragma unroll
      for (int off = 1; off < 16; off <<= 1)
        ls += __shfl_xor(ls, off, 32);
      l_i[r] = l_i[r] * alpha + ls;
      m_i[r] = m_new;
      #pragma unroll
      for (int n = 0; n < 8; ++n) o_acc[n][r] *= alpha;
      Pb[wave][sel * 8 + r][ln16]      = (_Float16)p0;
      Pb[wave][sel * 8 + r][16 + ln16] = (_Float16)p1;
    }

    // ---- reload P in A-layout (same-wave DS ops are in-order) ----
    v8h plo = *(const v8h*)&Pb[wave][ln16][sel * 8];
    v8h phi = *(const v8h*)&Pb[wave][ln16][16 + sel * 8];
    v16h pfrag = cat8h(plo, phi);

    // ---- preload all V fragments, then 8 PV WMMAs ----
    v16h vfr[8];
    #pragma unroll
    for (int n = 0; n < 8; ++n) {
      // B[k][nd] = Vt[n*16+nd][k]; per lane nd=ln16, k=sel*8+s / 16+sel*8+(s-8) (table order)
      v8h lo = *(const v8h*)&Vt[n * 16 + ln16][sel * 8];
      v8h hi = *(const v8h*)&Vt[n * 16 + ln16][16 + sel * 8];
      vfr[n] = cat8h(lo, hi);
    }
    #pragma unroll
    for (int n = 0; n < 8; ++n)
      o_acc[n] = __builtin_amdgcn_wmma_f32_16x16x32_f16(
          false, pfrag, false, vfr[n], (short)0, o_acc[n], false, false);

    nb ^= 1;
  }

  // ---- epilogue: normalize and store fp32 ----
  #pragma unroll
  for (int r = 0; r < 8; ++r) {
    const int row = q_row0 + sel * 8 + r;
    const float inv = 1.0f / l_i[r];
    float* op = O + ((size_t)(b * SEQ + row) * NUM_HEADS + h) * HEAD_DIM;
    #pragma unroll
    for (int n = 0; n < 8; ++n)
      op[n * 16 + ln16] = o_acc[n][r] * inv;
  }
}

extern "C" void kernel_launch(void* const* d_in, const int* in_sizes, int n_in,
                              void* d_out, int out_size, void* d_ws, size_t ws_size,
                              hipStream_t stream) {
  (void)in_sizes; (void)n_in; (void)out_size; (void)d_ws; (void)ws_size;
  const float* q = (const float*)d_in[0];
  const float* k = (const float*)d_in[1];
  const float* v = (const float*)d_in[2];
  float* o = (float*)d_out;
  dim3 grid(BATCH * NUM_HEADS * (SEQ / QTILE));  // 1024 blocks
  fa2_gqa_causal<<<grid, dim3(256), 0, stream>>>(q, k, v, o);
}
